// CO2Net_22488448762680
// MI455X (gfx1250) — matrix operand order repaired and checked by reading, not compiled, not run on hardware
//
#include <hip/hip_runtime.h>
#include <hip/hip_bf16.h>

// ---------------------------------------------------------------------------
// PointNet++ (part-seg) forward for MI455X / gfx1250.
// All MLP einsums -> f16 WMMA GEMMs (v_wmma_f32_16x16x32_f16), K padded to 32.
// GEMM: 4 waves/block, each wave computes a 16x32 strip (2 WMMA tiles, shared
// A fragment); double-buffered LDS fed by GLOBAL_LOAD_ASYNC_TO_LDS_B128 with
// s_wait_asynccnt pipelining; weights pre-transposed so both fragments are
// K-contiguous and load as ds_load_b128 pairs.
// ---------------------------------------------------------------------------

typedef __attribute__((ext_vector_type(16))) _Float16 v16h;
typedef __attribute__((ext_vector_type(8)))  _Float16 v8h;
typedef __attribute__((ext_vector_type(8)))  float    v8f;

#if defined(__has_builtin)
#if __has_builtin(__builtin_amdgcn_global_load_async_to_lds_b128)
#define ATHENA_ASYNC_LDS 1
#endif
#endif
#ifndef ATHENA_ASYNC_LDS
#define ATHENA_ASYNC_LDS 0
#endif

#if ATHENA_ASYNC_LDS
// Builtin signature (from compiler diagnostic): param0 is v4i* in AS(1).
typedef int v4i __attribute__((vector_size(16)));
typedef __attribute__((address_space(1))) v4i* as1_v4i_p;
typedef __attribute__((address_space(3))) v4i* as3_v4i_p;

__device__ __forceinline__ void async_copy16(const void* g, void* l) {
  // Low 32 bits of a generic LDS pointer are the LDS byte offset.
  __builtin_amdgcn_global_load_async_to_lds_b128(
      (as1_v4i_p)(uintptr_t)g,
      (as3_v4i_p)(unsigned int)(uintptr_t)l,
      0, 0);
}
__device__ __forceinline__ void async_wait0() {
#if __has_builtin(__builtin_amdgcn_s_wait_asynccnt)
  __builtin_amdgcn_s_wait_asynccnt(0);
#else
  asm volatile("s_wait_asynccnt 0" ::: "memory");
#endif
}
__device__ __forceinline__ void async_wait3() {
#if __has_builtin(__builtin_amdgcn_s_wait_asynccnt)
  __builtin_amdgcn_s_wait_asynccnt(3);
#else
  asm volatile("s_wait_asynccnt 3" ::: "memory");
#endif
}
#endif

// --------------------------- input transpose -------------------------------
// xyz (B,5,4096) -> pts (B,4096,5), l0xyz (B,4096,3)
__global__ void k_transpose_in(const float* __restrict__ xyz, float* __restrict__ pts,
                               float* __restrict__ l0, int B, int C, int Np) {
  int i = blockIdx.x * 256 + threadIdx.x;
  if (i >= B * Np) return;
  int b = i / Np, n = i % Np;
  for (int c = 0; c < C; ++c) {
    float v = xyz[((size_t)b * C + c) * Np + n];
    pts[(size_t)i * C + c] = v;
    if (c < 3) l0[(size_t)i * 3 + c] = v;
  }
}

// ------------------------------- FPS ---------------------------------------
__global__ void k_fps(const float* __restrict__ xyz, int* __restrict__ out,
                      int Np, int npoint) {
  __shared__ float dist[4096];
  __shared__ float rmax[256];
  __shared__ int   ridx[256];
  __shared__ int   sfar;
  int b = blockIdx.x, t = threadIdx.x;
  const float* base = xyz + (size_t)b * Np * 3;
  for (int i = t; i < Np; i += 256) dist[i] = 1e10f;
  if (t == 0) sfar = 0;
  __syncthreads();
  for (int it = 0; it < npoint; ++it) {
    int far = sfar;
    if (t == 0) out[b * npoint + it] = far;
    float cx = base[far * 3 + 0], cy = base[far * 3 + 1], cz = base[far * 3 + 2];
    float bm = -1.0f; int bi = 0;
    for (int i = t; i < Np; i += 256) {
      float dx = base[i * 3 + 0] - cx;
      float dy = base[i * 3 + 1] - cy;
      float dz = base[i * 3 + 2] - cz;
      float d = dx * dx + dy * dy + dz * dz;
      float nd = fminf(dist[i], d);
      dist[i] = nd;
      if (nd > bm) { bm = nd; bi = i; }
    }
    rmax[t] = bm; ridx[t] = bi;
    __syncthreads();
    for (int s = 128; s > 0; s >>= 1) {
      if (t < s && rmax[t + s] > rmax[t]) { rmax[t] = rmax[t + s]; ridx[t] = ridx[t + s]; }
      __syncthreads();
    }
    if (t == 0) sfar = ridx[0];
    __syncthreads();
  }
}

// gather 3-vectors by index: dst[i] = src[b, idx[i]]
__global__ void k_gather3(const float* __restrict__ src, const int* __restrict__ idx,
                          float* __restrict__ dst, int B, int Np, int S) {
  int i = blockIdx.x * 256 + threadIdx.x;
  if (i >= B * S) return;
  int b = i / S;
  int j = idx[i];
  const float* p = src + ((size_t)b * Np + j) * 3;
  dst[3 * i + 0] = p[0]; dst[3 * i + 1] = p[1]; dst[3 * i + 2] = p[2];
}

// --------------------------- ball query (K=32) -----------------------------
__global__ void k_ball_query(const float* __restrict__ xyz, const float* __restrict__ new_xyz,
                             int* __restrict__ gidx, int B, int Np, int S, float r2) {
  int i = blockIdx.x * 256 + threadIdx.x;
  if (i >= B * S) return;
  int b = i / S;
  const float* p = new_xyz + (size_t)i * 3;
  float px = p[0], py = p[1], pz = p[2];
  const float* q = xyz + (size_t)b * Np * 3;
  int* o = gidx + (size_t)i * 32;
  int cnt = 0;
  for (int j = 0; j < Np && cnt < 32; ++j) {
    float dx = q[3 * j + 0] - px, dy = q[3 * j + 1] - py, dz = q[3 * j + 2] - pz;
    if (dx * dx + dy * dy + dz * dz <= r2) o[cnt++] = j;
  }
  int f = cnt > 0 ? o[0] : 0;
  for (int k = cnt; k < 32; ++k) o[k] = f;
}

// gather grouped (xyz-diff ++ feats) directly into padded f16 GEMM operand
__global__ void k_group_gather(const float* __restrict__ xyz, const float* __restrict__ feats,
                               const float* __restrict__ new_xyz, const int* __restrict__ gidx,
                               _Float16* __restrict__ out, int B, int Np, int Cf, int S, int Cpad) {
  int i = blockIdx.x * 256 + threadIdx.x;
  int total = B * S * 32;
  if (i >= total) return;
  int s = (i >> 5) % S;
  int b = i / (S * 32);
  int j = gidx[i];
  const float* nx = new_xyz + ((size_t)b * S + s) * 3;
  const float* px = xyz + ((size_t)b * Np + j) * 3;
  _Float16* o = out + (size_t)i * Cpad;
  o[0] = (_Float16)(px[0] - nx[0]);
  o[1] = (_Float16)(px[1] - nx[1]);
  o[2] = (_Float16)(px[2] - nx[2]);
  const float* f = feats + ((size_t)b * Np + j) * Cf;
  for (int c = 0; c < Cf; ++c) o[3 + c] = (_Float16)f[c];
  for (int c = 3 + Cf; c < Cpad; ++c) o[c] = (_Float16)0.f;
}

// weight fp32 (Cin,Cout) -> TRANSPOSED f16 (Cout,Cpad), zero-padded K tail
__global__ void k_wtof16t(const float* __restrict__ W, _Float16* __restrict__ Wt,
                          int Cin, int Cpad, int Cout) {
  int i = blockIdx.x * 256 + threadIdx.x;
  if (i >= Cout * Cpad) return;
  int c = i / Cpad;           // output channel (row of Wt)
  int r = i % Cpad;           // input channel  (col of Wt)
  Wt[i] = (r < Cin) ? (_Float16)W[r * Cout + c] : (_Float16)0.f;
}

// activations fp32 (M,Cin) -> f16 (M,Cpad)
__global__ void k_tof16(const float* __restrict__ in, _Float16* __restrict__ out,
                        size_t M, int Cin, int Cpad) {
  size_t i = (size_t)blockIdx.x * 256 + threadIdx.x;
  if (i >= M * (size_t)Cpad) return;
  int c = (int)(i % Cpad);
  size_t r = i / Cpad;
  out[i] = (c < Cin) ? (_Float16)in[r * Cin + c] : (_Float16)0.f;
}

// ------------------------------ WMMA GEMM ----------------------------------
// C[M,N] = A[M,Kp] (f16 row-major) x W (f16, stored transposed as Wt[N,Kp]).
// Block = 4 waves (128 thr). Block tile 64x32; wave w owns rows [16w,16w+16)
// and both 16-wide N tiles (A fragment reused across 2 WMMAs).
// Double-buffered LDS; async copies for tile k+1 overlap WMMA on tile k.
__global__ void k_wmma_gemm(const _Float16* __restrict__ A, const _Float16* __restrict__ Wt,
                            const float* __restrict__ bias, float* __restrict__ C,
                            int M, int N, int Kp) {
  __shared__ __align__(16) _Float16 As[2][64][32];   // [buf][m][k]
  __shared__ __align__(16) _Float16 Bs[2][32][32];   // [buf][n][k]  (K-contiguous)
  int tid  = threadIdx.x;          // 0..127
  int wave = tid >> 5;
  int lane = tid & 31;
  int row16 = lane & 15;
  int hi    = lane >> 4;
  int m0 = blockIdx.x * 64;
  int n0 = blockIdx.y * 32;

  // Per-stage: every thread issues exactly 3 async copies (2 A chunks + 1 B chunk).
  auto stage = [&](int b, int k0) {
#if ATHENA_ASYNC_LDS
#pragma unroll
    for (int j = 0; j < 2; ++j) {                    // 256 x 16B chunks of A
      int idx = tid + j * 128;
      int r = idx >> 2, kc = (idx & 3) * 8;
      async_copy16(A + (size_t)(m0 + r) * Kp + k0 + kc, &As[b][r][kc]);
    }
    {                                                // 128 x 16B chunks of B
      int r = tid >> 2, kc = (tid & 3) * 8;
      async_copy16(Wt + (size_t)(n0 + r) * Kp + k0 + kc, &Bs[b][r][kc]);
    }
#else
#pragma unroll
    for (int j = 0; j < 2; ++j) {
      int idx = tid + j * 128;
      int r = idx >> 2, kc = (idx & 3) * 8;
      const _Float16* g = A + (size_t)(m0 + r) * Kp + k0 + kc;
      _Float16* d = &As[b][r][kc];
#pragma unroll
      for (int u = 0; u < 8; ++u) d[u] = g[u];
    }
    {
      int r = tid >> 2, kc = (tid & 3) * 8;
      const _Float16* g = Wt + (size_t)(n0 + r) * Kp + k0 + kc;
      _Float16* d = &Bs[b][r][kc];
#pragma unroll
      for (int u = 0; u < 8; ++u) d[u] = g[u];
    }
#endif
  };

  v8f acc0 = {}, acc1 = {};
  stage(0, 0);
  for (int k0 = 0; k0 < Kp; k0 += 32) {
    int cur = (k0 >> 5) & 1;
    bool more = (k0 + 32) < Kp;
    if (more) {
      stage(cur ^ 1, k0 + 32);
      __builtin_prefetch(A + (size_t)(m0 + (tid >> 1)) * Kp + k0 + 64, 0, 0);
#if ATHENA_ASYNC_LDS
      async_wait3();     // FIFO: <=3 outstanding => current tile's copies done
    } else {
      async_wait0();
#endif
    }
    __syncthreads();     // current buffer visible to all waves

    // ---- fragments (contiguous K = 16*hi + s, same order for A and B), all ds_load_b128 pairs ---------
    // A 16x32: lane(row16,hi) -> K [16*hi..16*hi+16)
    union { v16h v; v8h h[2]; } ua, ub0, ub1;
    const _Float16* ar = &As[cur][wave * 16 + row16][0];
    ua.h[0] = *(const v8h*)(ar + 16 * hi);
    ua.h[1] = *(const v8h*)(ar + 16 * hi + 8);
    // B 32x16 per tile: lanes0-15 K=0..15, lanes16-31 K=16..31, col=row16
    const _Float16* br0 = &Bs[cur][row16][0];        // N tile 0
    const _Float16* br1 = &Bs[cur][16 + row16][0];   // N tile 1
    ub0.h[0] = *(const v8h*)(br0 + 16 * hi);
    ub0.h[1] = *(const v8h*)(br0 + 16 * hi + 8);
    ub1.h[0] = *(const v8h*)(br1 + 16 * hi);
    ub1.h[1] = *(const v8h*)(br1 + 16 * hi + 8);

    acc0 = __builtin_amdgcn_wmma_f32_16x16x32_f16(false, ua.v, false, ub0.v,
                                                  (short)0, acc0, false, false);
    acc1 = __builtin_amdgcn_wmma_f32_16x16x32_f16(false, ua.v, false, ub1.v,
                                                  (short)0, acc1, false, false);
    __syncthreads();     // protect 'cur' buffer before it is re-staged
  }

  // ---- epilogue: VGPR g holds (M = g + 8*hi, N = col) ---------------------
  float b0 = bias ? bias[n0 + row16] : 0.f;
  float b1 = bias ? bias[n0 + 16 + row16] : 0.f;
#pragma unroll
  for (int g = 0; g < 8; ++g) {
    int r = m0 + wave * 16 + g + 8 * hi;
    C[(size_t)r * N + n0 + row16]      = acc0[g] + b0;
    C[(size_t)r * N + n0 + 16 + row16] = acc1[g] + b1;
  }
}

// ----------------------- batchnorm (batch stats) ---------------------------
__global__ void k_bn_stats(const float* __restrict__ X, float* __restrict__ stats,
                           int M, int N) {
  __shared__ float s1[256], s2[256];
  int c = blockIdx.x, t = threadIdx.x;
  float a = 0.f, b = 0.f;
  for (int r = t; r < M; r += 256) {
    float v = X[(size_t)r * N + c];
    a += v; b += v * v;
  }
  s1[t] = a; s2[t] = b;
  __syncthreads();
  for (int s = 128; s > 0; s >>= 1) {
    if (t < s) { s1[t] += s1[t + s]; s2[t] += s2[t + s]; }
    __syncthreads();
  }
  if (t == 0) {
    float m = s1[0] / (float)M;
    stats[c]     = m;
    stats[N + c] = fmaxf(s2[0] / (float)M - m * m, 0.f);
  }
}

__global__ void k_bn_apply(float* __restrict__ X, const float* __restrict__ stats,
                           const float* __restrict__ g, const float* __restrict__ be,
                           size_t M, int N, int relu) {
  size_t i = (size_t)blockIdx.x * 256 + threadIdx.x;
  if (i >= M * (size_t)N) return;
  int c = (int)(i % N);
  float y = (X[i] - stats[c]) * rsqrtf(stats[N + c] + 1e-5f) * g[c] + be[c];
  if (relu) y = fmaxf(y, 0.f);
  X[i] = y;
}

// ------------------------------ max pool over K=32 -------------------------
__global__ void k_maxpool(const float* __restrict__ h, float* __restrict__ out,
                          int rows, int C) {
  int i = blockIdx.x * 256 + threadIdx.x;
  if (i >= rows * C) return;
  int c = i % C;
  size_t r = (size_t)(i / C);
  const float* p = h + r * 32 * (size_t)C + c;
  float m = p[0];
  for (int k = 1; k < 32; ++k) m = fmaxf(m, p[(size_t)k * C]);
  out[i] = m;
}

// ------------------- 3-NN inverse-distance interpolation -------------------
__global__ void k_three_interp(const float* __restrict__ xyz1, const float* __restrict__ xyz2,
                               const float* __restrict__ pts2, float* __restrict__ out,
                               int B, int N1, int N2, int C2, int co, int ldo) {
  int i = blockIdx.x * 256 + threadIdx.x;
  if (i >= B * N1) return;
  int b = i / N1;
  const float* p = xyz1 + (size_t)i * 3;
  float px = p[0], py = p[1], pz = p[2];
  float d0 = 1e30f, d1 = 1e30f, d2 = 1e30f;
  int i0 = 0, i1 = 0, i2 = 0;
  const float* q = xyz2 + (size_t)b * N2 * 3;
  for (int j = 0; j < N2; ++j) {
    float dx = q[3 * j + 0] - px, dy = q[3 * j + 1] - py, dz = q[3 * j + 2] - pz;
    float d = dx * dx + dy * dy + dz * dz;
    if (d < d0)      { d2 = d1; i2 = i1; d1 = d0; i1 = i0; d0 = d; i0 = j; }
    else if (d < d1) { d2 = d1; i2 = i1; d1 = d;  i1 = j; }
    else if (d < d2) { d2 = d;  i2 = j; }
  }
  float w0 = 1.f / (d0 + 1e-8f), w1 = 1.f / (d1 + 1e-8f), w2 = 1.f / (d2 + 1e-8f);
  float ws = w0 + w1 + w2;
  w0 /= ws; w1 /= ws; w2 /= ws;
  const float* f = pts2 + (size_t)b * N2 * C2;
  float* o = out + (size_t)i * ldo + co;
  for (int c = 0; c < C2; ++c)
    o[c] = w0 * f[(size_t)i0 * C2 + c] + w1 * f[(size_t)i1 * C2 + c] + w2 * f[(size_t)i2 * C2 + c];
}

// concat-copy points1 into the left columns of the fp input buffer
__global__ void k_copy2d(const float* __restrict__ in, float* __restrict__ out,
                         size_t rows, int C, int ldo, int co) {
  size_t i = (size_t)blockIdx.x * 256 + threadIdx.x;
  if (i >= rows * (size_t)C) return;
  int c = (int)(i % C);
  size_t r = i / C;
  out[r * ldo + co + c] = in[i];
}

// conv2: (B*N,128) x (128,1) + b
__global__ void k_conv2(const float* __restrict__ h, const float* __restrict__ W,
                        const float* __restrict__ b, float* __restrict__ out,
                        int total, int C) {
  int i = blockIdx.x * 256 + threadIdx.x;
  if (i >= total) return;
  const float* p = h + (size_t)i * C;
  float s = b[0];
  for (int c = 0; c < C; ++c) s += p[c] * W[c];
  out[i] = s;
}

// l4 (B,16,512) -> out (B,512,16)
__global__ void k_l4out(const float* __restrict__ l4, float* __restrict__ out, int B) {
  int i = blockIdx.x * 256 + threadIdx.x;
  int total = B * 512 * 16;
  if (i >= total) return;
  int s = i % 16;
  int c = (i / 16) % 512;
  int b = i / (512 * 16);
  out[i] = l4[((size_t)b * 16 + s) * 512 + c];
}

// ---------------------------------------------------------------------------
extern "C" void kernel_launch(void* const* d_in, const int* in_sizes, int n_in,
                              void* d_out, int out_size, void* d_ws, size_t ws_size,
                              hipStream_t stream) {
  (void)in_sizes; (void)n_in; (void)out_size; (void)ws_size;
  const int B = 8, Np0 = 4096;

  // ---- workspace slab allocator -----------------------------------------
  char* wsb = (char*)d_ws;
  size_t off = 0;
  auto alloc = [&](size_t bytes) -> void* {
    off = (off + 255) & ~(size_t)255;
    void* p = wsb + off;
    off += bytes;
    return p;
  };

  float* pts   = (float*)alloc((size_t)B * Np0 * 5 * 4);
  float* l0xyz = (float*)alloc((size_t)B * Np0 * 3 * 4);
  int SL[5] = {4096, 1024, 256, 64, 16};
  float* xyzL[5];
  xyzL[0] = l0xyz;
  for (int l = 1; l < 5; ++l) xyzL[l] = (float*)alloc((size_t)B * SL[l] * 3 * 4);
  int* fidx = (int*)alloc((size_t)B * 1024 * 4);
  int* gidx = (int*)alloc((size_t)B * 1024 * 32 * 4);

  _Float16* h16a = (_Float16*)alloc((size_t)18 << 20);
  _Float16* h16b = (_Float16*)alloc((size_t)18 << 20);
  float*    f32a = (float*)alloc((size_t)68 << 20);   // max 262144x64 fp32
  float*    f32c = (float*)alloc((size_t)20 << 20);   // fp concat input
  _Float16* wh   = (_Float16*)alloc((size_t)768 * 512 * 2);
  float*    stats = (float*)alloc((size_t)2 * 512 * 4);

  float* l1p = (float*)alloc((size_t)B * 1024 * 64 * 4);
  float* l2p = (float*)alloc((size_t)B * 256 * 128 * 4);
  float* l3p = (float*)alloc((size_t)B * 64 * 256 * 4);
  float* l4p = (float*)alloc((size_t)B * 16 * 512 * 4);
  float* n3  = (float*)alloc((size_t)B * 64 * 256 * 4);
  float* n2  = (float*)alloc((size_t)B * 256 * 256 * 4);
  float* n1  = (float*)alloc((size_t)B * 1024 * 128 * 4);
  float* n0  = (float*)alloc((size_t)B * 4096 * 128 * 4);

  // ---- param stream ------------------------------------------------------
  int pi = 1;
  auto nextp = [&]() -> const float* { return (const float*)d_in[pi++]; };
  auto pad32 = [](int c) { return (c + 31) & ~31; };

  auto tof16 = [&](const float* F, _Float16* H, size_t M, int Cin, int Cpad) {
    size_t t = M * (size_t)Cpad;
    k_tof16<<<(unsigned)((t + 255) / 256), 256, 0, stream>>>(F, H, M, Cin, Cpad);
  };

  // GEMM + bias + batchnorm(+relu); A is padded f16 (M x Cpad)
  auto gemm = [&](const _Float16* Ah, int M, int Cin, int Cpad,
                  const float* W, const float* bias, const float* g, const float* be,
                  int Cout, float* outF, bool relu) {
    int tw = Cout * Cpad;
    k_wtof16t<<<(tw + 255) / 256, 256, 0, stream>>>(W, wh, Cin, Cpad, Cout);
    dim3 grid((unsigned)(M / 64), (unsigned)(Cout / 32));
    k_wmma_gemm<<<grid, 128, 0, stream>>>(Ah, wh, bias, outF, M, Cout, Cpad);
    k_bn_stats<<<Cout, 256, 0, stream>>>(outF, stats, M, Cout);
    size_t t = (size_t)M * Cout;
    k_bn_apply<<<(unsigned)((t + 255) / 256), 256, 0, stream>>>(outF, stats, g, be,
                                                                (size_t)M, Cout, relu ? 1 : 0);
  };

  // ---- set-abstraction stage --------------------------------------------
  auto sa = [&](const float* xyz_in, int Npts, const float* feats, int Cf,
                int S, float radius, float* pooled, float* new_xyz, const int widths[3]) {
    k_fps<<<B, 256, 0, stream>>>(xyz_in, fidx, Npts, S);
    k_gather3<<<(B * S + 255) / 256, 256, 0, stream>>>(xyz_in, fidx, new_xyz, B, Npts, S);
    k_ball_query<<<(B * S + 255) / 256, 256, 0, stream>>>(xyz_in, new_xyz, gidx, B, Npts, S,
                                                          radius * radius);
    int Cin = 3 + Cf, Cpad = pad32(Cin);
    int M = B * S * 32;
    k_group_gather<<<(M + 255) / 256, 256, 0, stream>>>(xyz_in, feats, new_xyz, gidx, h16a,
                                                        B, Npts, Cf, S, Cpad);
    const _Float16* cur = h16a;
    _Float16* nxt = h16b;
    int cIn = Cin, cPad = Cpad;
    for (int l = 0; l < 3; ++l) {
      const float* W = nextp(); const float* bb = nextp();
      const float* gg = nextp(); const float* ee = nextp();
      int Cout = widths[l];
      gemm(cur, M, cIn, cPad, W, bb, gg, ee, Cout, f32a, true);
      if (l < 2) {
        tof16(f32a, nxt, (size_t)M, Cout, pad32(Cout));
        cur = nxt;
        nxt = (nxt == h16b) ? h16a : h16b;
        cIn = Cout; cPad = pad32(Cout);
      }
    }
    int tp = B * S * widths[2];
    k_maxpool<<<(tp + 255) / 256, 256, 0, stream>>>(f32a, pooled, B * S, widths[2]);
  };

  // ---- feature-propagation stage ----------------------------------------
  auto fp = [&](const float* xyz1, int N1, const float* xyz2, int N2,
                const float* p1, int C1, const float* p2, int C2,
                float* outF, const int* widths, int nlayers) {
    int Cin = C1 + C2;
    if (p1) {
      size_t t = (size_t)B * N1 * C1;
      k_copy2d<<<(unsigned)((t + 255) / 256), 256, 0, stream>>>(p1, f32c, (size_t)B * N1,
                                                                C1, Cin, 0);
    }
    k_three_interp<<<(B * N1 + 255) / 256, 256, 0, stream>>>(xyz1, xyz2, p2, f32c,
                                                             B, N1, N2, C2, C1, Cin);
    int M = B * N1, cIn = Cin, cPad = pad32(Cin);
    tof16(f32c, h16a, (size_t)M, Cin, cPad);
    const _Float16* cur = h16a;
    _Float16* nxt = h16b;
    for (int l = 0; l < nlayers; ++l) {
      const float* W = nextp(); const float* bb = nextp();
      const float* gg = nextp(); const float* ee = nextp();
      int Cout = widths[l];
      float* dst = (l == nlayers - 1) ? outF : f32a;
      gemm(cur, M, cIn, cPad, W, bb, gg, ee, Cout, dst, true);
      if (l < nlayers - 1) {
        tof16(f32a, nxt, (size_t)M, Cout, pad32(Cout));
        cur = nxt;
        nxt = (nxt == h16b) ? h16a : h16b;
        cIn = Cout; cPad = pad32(Cout);
      }
    }
  };

  // ---- forward graph -----------------------------------------------------
  k_transpose_in<<<(B * Np0 + 255) / 256, 256, 0, stream>>>((const float*)d_in[0], pts, l0xyz,
                                                            B, 5, Np0);
  const int w1[3] = {32, 32, 64},  w2[3] = {64, 64, 128};
  const int w3[3] = {128, 128, 256}, w4[3] = {256, 256, 512};
  sa(l0xyz,  4096, pts, 5,   1024, 0.1f, l1p, xyzL[1], w1);
  sa(xyzL[1], 1024, l1p, 64,  256, 0.2f, l2p, xyzL[2], w2);
  sa(xyzL[2],  256, l2p, 128,  64, 0.4f, l3p, xyzL[3], w3);
  sa(xyzL[3],   64, l3p, 256,  16, 0.8f, l4p, xyzL[4], w4);

  const int f4[2] = {256, 256}, f3[2] = {256, 256}, f2[2] = {256, 128}, f1[3] = {128, 128, 128};
  fp(xyzL[3],  64, xyzL[4],  16, l3p, 256, l4p, 512, n3, f4, 2);
  fp(xyzL[2], 256, xyzL[3],  64, l2p, 128, n3, 256, n2, f3, 2);
  fp(xyzL[1], 1024, xyzL[2], 256, l1p, 64, n2, 256, n1, f2, 2);
  fp(l0xyz,  4096, xyzL[1], 1024, nullptr, 0, n1, 128, n0, f1, 3);

  // conv1 (+bn+relu) then conv2 head
  {
    const float* W = nextp(); const float* bb = nextp();
    const float* gg = nextp(); const float* ee = nextp();
    tof16(n0, h16a, (size_t)B * 4096, 128, 128);
    gemm(h16a, B * 4096, 128, 128, W, bb, gg, ee, 128, f32a, true);
  }
  {
    const float* W = nextp(); const float* bb = nextp();
    k_conv2<<<(B * 4096 + 255) / 256, 256, 0, stream>>>(f32a, W, bb, (float*)d_out,
                                                        B * 4096, 128);
  }
  k_l4out<<<(B * 512 * 16 + 255) / 256, 256, 0, stream>>>(l4p, (float*)d_out + B * 4096, B);
}
